// LinearTernary_41566693491251
// MI455X (gfx1250) — compile-verified, matrix operand order checked by reading, not run on hardware
//
#include <hip/hip_runtime.h>
#include <hip/hip_bf16.h>
#include <stdint.h>

// ---------------------------------------------------------------------------
// LinearTernary on MI455X (gfx1250, wave32, WMMA)
//
//   out[M,N] = x[M,K] @ ternarize(w)[N,K]^T + bias[N]
//   M=8192 (tokens), K=4096 (in), N=16384 (out)
//
//   1) pre-pass A: x (f32) -> x_hi, x_lo (bf16 pair, x ~= hi + lo, ~17 bits)
//   2) pre-pass B: w (f32) -> ternary {-1,0,+1} stored as exact bf16
//   3) GEMM: block tile 128x128x32, 8 waves, each wave 4x2 tiles of
//      v_wmma_f32_16x16x32_bf16 (x2 for hi/lo), double-buffered LDS with
//      async global->LDS staging (ASYNCcnt) overlapping the next tile fetch
//      with the current tile's WMMAs.
//
// Workspace layout in d_ws (needs (2*M*K + N*K)*2 = 256 MB):
//   [0)      x_hi bf16 M*K   [M*K) x_lo bf16 M*K   [2*M*K) w_t bf16 N*K
// ---------------------------------------------------------------------------

typedef __attribute__((ext_vector_type(16))) __bf16 v16bf;
typedef __attribute__((ext_vector_type(8)))  float  v8f;

constexpr int M = 8192;    // TOKENS
constexpr int K = 4096;    // IN_FEATURES
constexpr int N = 16384;   // OUT_FEATURES

constexpr int BM = 128;
constexpr int BN = 128;
constexpr int BK = 32;

// ---- CDNA5 async global->LDS path (guarded so compile never regresses) ----
#if defined(__gfx1250__) && \
    __has_builtin(__builtin_amdgcn_global_load_async_to_lds_b128)
#define USE_ASYNC_LDS 1
#else
#define USE_ASYNC_LDS 0
#endif

#if USE_ASYNC_LDS
// Match the builtin's parameter types exactly:
//   src: 'int __attribute__((vector_size(16))) __device__ *'   (addrspace(1))
//   dst: same vector type in LDS                               (addrspace(3))
typedef int b128_t __attribute__((vector_size(4 * sizeof(int))));
typedef __attribute__((address_space(1))) b128_t glb_b128;
typedef __attribute__((address_space(3))) b128_t lds_b128;

static __device__ __forceinline__ void async_copy16(const void* g, void* l) {
  // global_load_async_to_lds_b128: 16B/lane, LDS write bypasses VGPRs.
  __builtin_amdgcn_global_load_async_to_lds_b128((glb_b128*)g, (lds_b128*)l,
                                                 /*imm offset*/ 0,
                                                 /*cpol*/ 0);
}

static __device__ __forceinline__ void wait_async_zero() {
#if __has_builtin(__builtin_amdgcn_s_wait_asynccnt)
  __builtin_amdgcn_s_wait_asynccnt(0);
#else
  asm volatile("s_wait_asynccnt 0x0" ::: "memory");
#endif
}
#endif  // USE_ASYNC_LDS

static __device__ __forceinline__ uint16_t bf16_rne_bits(float f) {
  uint32_t u = __builtin_bit_cast(uint32_t, f);
  uint32_t r = u + 0x7FFFu + ((u >> 16) & 1u);  // round-to-nearest-even
  return (uint16_t)(r >> 16);
}

// ---------------------------------------------------------------------------
// Pre-pass A: split x into bf16 hi + bf16 lo (4 elements / thread, float4 IO)
// ---------------------------------------------------------------------------
__global__ __launch_bounds__(256)
void split_x_bf16_kernel(const float* __restrict__ x,
                         uint16_t* __restrict__ xh,
                         uint16_t* __restrict__ xl) {
  const size_t i = ((size_t)blockIdx.x * blockDim.x + threadIdx.x) * 4;
  const float4 v = *(const float4*)(x + i);
  float e[4] = {v.x, v.y, v.z, v.w};
  uint32_t hp[2], lp[2];
#pragma unroll
  for (int p = 0; p < 2; ++p) {
    uint32_t h0, h1, l0, l1;
    {
      uint16_t hb = bf16_rne_bits(e[2 * p + 0]);
      float hf = __builtin_bit_cast(float, (uint32_t)hb << 16);
      h0 = hb;
      l0 = bf16_rne_bits(e[2 * p + 0] - hf);
    }
    {
      uint16_t hb = bf16_rne_bits(e[2 * p + 1]);
      float hf = __builtin_bit_cast(float, (uint32_t)hb << 16);
      h1 = hb;
      l1 = bf16_rne_bits(e[2 * p + 1] - hf);
    }
    hp[p] = h0 | (h1 << 16);
    lp[p] = l0 | (l1 << 16);
  }
  *(uint2*)(xh + i) = make_uint2(hp[0], hp[1]);
  *(uint2*)(xl + i) = make_uint2(lp[0], lp[1]);
}

// ---------------------------------------------------------------------------
// Pre-pass B: ternarize w -> exact bf16 {-1, 0, +1}
// ---------------------------------------------------------------------------
__global__ __launch_bounds__(256)
void ternarize_w_bf16_kernel(const float* __restrict__ w,
                             uint16_t* __restrict__ wt) {
  const size_t i = ((size_t)blockIdx.x * blockDim.x + threadIdx.x) * 4;
  const float4 v = *(const float4*)(w + i);
  const float e[4] = {v.x, v.y, v.z, v.w};
  uint32_t t[4];
#pragma unroll
  for (int j = 0; j < 4; ++j) {
    uint32_t u = __builtin_bit_cast(uint32_t, e[j]);
    uint32_t mag = u & 0x7FFFFFFFu;
    t[j] = (mag < 0x3F000000u) ? 0u : (0x3F80u | ((u >> 16) & 0x8000u));
  }
  *(uint2*)(wt + i) = make_uint2(t[0] | (t[1] << 16), t[2] | (t[3] << 16));
}

// ---------------------------------------------------------------------------
// WMMA GEMM: C = A_hi*B + A_lo*B (+bias)
// ---------------------------------------------------------------------------
static __device__ __forceinline__ v16bf make_frag(uint4 q0, uint4 q1) {
  union { uint4 q[2]; v16bf v; } u;
  u.q[0] = q0;
  u.q[1] = q1;
  return u.v;
}

__global__ __launch_bounds__(256)
void ternary_gemm_wmma_kernel(const uint16_t* __restrict__ xh,
                              const uint16_t* __restrict__ xl,
                              const uint16_t* __restrict__ wt,
                              const float* __restrict__ bias,
                              float* __restrict__ out) {
  __shared__ uint16_t Ah[2][BM][BK];  // 2 x 8 KB
  __shared__ uint16_t Al[2][BM][BK];  // 2 x 8 KB
  __shared__ uint16_t Bs[2][BN][BK];  // 2 x 8 KB   (rows = n, cols = k)

  const int tid  = threadIdx.x;
  const int lane = tid & 31;
  const int wid  = tid >> 5;  // 0..7
  const int wm   = wid & 1;   // -> 64 M-rows per wave
  const int wn   = wid >> 1;  // -> 32 N-cols per wave

  const int m0 = blockIdx.y * BM;
  const int n0 = blockIdx.x * BN;

  // staging coordinates: 512 x 16B chunks per tile, 2 per thread
  const int r0 = tid >> 2;
  const int c0 = (tid & 3) << 3;
  const int r1 = (tid + 256) >> 2;
  const int c1 = ((tid + 256) & 3) << 3;

  auto stage = [&](int buf, int kt) {
#if USE_ASYNC_LDS
    async_copy16(&xh[(size_t)(m0 + r0) * K + kt + c0], &Ah[buf][r0][c0]);
    async_copy16(&xh[(size_t)(m0 + r1) * K + kt + c1], &Ah[buf][r1][c1]);
    async_copy16(&xl[(size_t)(m0 + r0) * K + kt + c0], &Al[buf][r0][c0]);
    async_copy16(&xl[(size_t)(m0 + r1) * K + kt + c1], &Al[buf][r1][c1]);
    async_copy16(&wt[(size_t)(n0 + r0) * K + kt + c0], &Bs[buf][r0][c0]);
    async_copy16(&wt[(size_t)(n0 + r1) * K + kt + c1], &Bs[buf][r1][c1]);
#else
    *(uint4*)&Ah[buf][r0][c0] = *(const uint4*)&xh[(size_t)(m0 + r0) * K + kt + c0];
    *(uint4*)&Ah[buf][r1][c1] = *(const uint4*)&xh[(size_t)(m0 + r1) * K + kt + c1];
    *(uint4*)&Al[buf][r0][c0] = *(const uint4*)&xl[(size_t)(m0 + r0) * K + kt + c0];
    *(uint4*)&Al[buf][r1][c1] = *(const uint4*)&xl[(size_t)(m0 + r1) * K + kt + c1];
    *(uint4*)&Bs[buf][r0][c0] = *(const uint4*)&wt[(size_t)(n0 + r0) * K + kt + c0];
    *(uint4*)&Bs[buf][r1][c1] = *(const uint4*)&wt[(size_t)(n0 + r1) * K + kt + c1];
#endif
  };

  // ISA 7.12.2 lane decomposition: lanes 0-15 carry K chunks {0..7,16..23},
  // lanes 16-31 carry {8..15,24..31}; M/N index = lane & 15.
  const int half = lane >> 4;
  const int lr   = lane & 15;
  const int cb   = half * 8;

  v8f acc[4][2] = {};

  constexpr int NIT = K / BK;  // 128
  stage(0, 0);                 // prologue fetch of tile 0

  for (int it = 0; it < NIT; ++it) {
    const int cur = it & 1;
    const int kt  = it * BK;

#if USE_ASYNC_LDS
    wait_async_zero();  // my async writes for buf[cur] have landed in LDS
#endif
    __syncthreads();    // everyone's writes visible; prev reads of buf[cur^1] done

    if (it + 1 < NIT) stage(cur ^ 1, kt + BK);  // overlap fetch with compute

    // ---- fragment loads from LDS -----------------------------------------
    v16bf afh[4], afl[4], bfr[2];
#pragma unroll
    for (int mt = 0; mt < 4; ++mt) {
      const uint16_t* ph = &Ah[cur][wm * 64 + mt * 16 + lr][cb];
      afh[mt] = make_frag(*(const uint4*)ph, *(const uint4*)(ph + 16));
      const uint16_t* pl = &Al[cur][wm * 64 + mt * 16 + lr][cb];
      afl[mt] = make_frag(*(const uint4*)pl, *(const uint4*)(pl + 16));
    }
#pragma unroll
    for (int nt = 0; nt < 2; ++nt) {
      const uint16_t* pb = &Bs[cur][wn * 32 + nt * 16 + lr][cb];
      bfr[nt] = make_frag(*(const uint4*)pb, *(const uint4*)(pb + 16));
    }

    // ---- 16 WMMAs per K-step per wave ------------------------------------
#pragma unroll
    for (int mt = 0; mt < 4; ++mt) {
#pragma unroll
      for (int nt = 0; nt < 2; ++nt) {
        acc[mt][nt] = __builtin_amdgcn_wmma_f32_16x16x32_bf16(
            false, afh[mt], false, bfr[nt], (short)0, acc[mt][nt],
            false, false);
        acc[mt][nt] = __builtin_amdgcn_wmma_f32_16x16x32_bf16(
            false, afl[mt], false, bfr[nt], (short)0, acc[mt][nt],
            false, false);
      }
    }
  }

  // ---- epilogue: C/D layout (ISA 7.12.2): VGPR v, lane l ->
  //      row = v + 8*(l>>4), col = l&15 ------------------------------------
#pragma unroll
  for (int mt = 0; mt < 4; ++mt) {
#pragma unroll
    for (int nt = 0; nt < 2; ++nt) {
      const int col = n0 + wn * 32 + nt * 16 + lr;
      const float b = bias[col];
      const int rbase = m0 + wm * 64 + mt * 16 + half * 8;
#pragma unroll
      for (int v = 0; v < 8; ++v) {
        out[(size_t)(rbase + v) * N + col] = acc[mt][nt][v] + b;
      }
    }
  }
}

// ---------------------------------------------------------------------------
extern "C" void kernel_launch(void* const* d_in, const int* in_sizes, int n_in,
                              void* d_out, int out_size, void* d_ws,
                              size_t ws_size, hipStream_t stream) {
  const float* x    = (const float*)d_in[0];  // [M, K] f32
  const float* w    = (const float*)d_in[1];  // [N, K] f32
  const float* bias = (const float*)d_in[2];  // [N]    f32
  float* out        = (float*)d_out;          // [M, N] f32

  uint16_t* xh = (uint16_t*)d_ws;             // bf16 hi  [M*K]
  uint16_t* xl = xh + (size_t)M * K;          // bf16 lo  [M*K]
  uint16_t* wt = xl + (size_t)M * K;          // bf16 w_t [N*K]

  split_x_bf16_kernel<<<((size_t)M * K / 4) / 256, 256, 0, stream>>>(x, xh, xl);
  ternarize_w_bf16_kernel<<<((size_t)N * K / 4) / 256, 256, 0, stream>>>(w, wt);

  dim3 grid(N / BN, M / BM);  // 128 x 64 blocks
  ternary_gemm_wmma_kernel<<<grid, 256, 0, stream>>>(xh, xl, wt, bias, out);
}
